// FrameTransformer_57638461112369
// MI455X (gfx1250) — compile-verified, matrix operand order checked by reading, not run on hardware
//
#include <hip/hip_runtime.h>
#include <cmath>

typedef __attribute__((ext_vector_type(16))) _Float16 v16h;
typedef __attribute__((ext_vector_type(4)))  _Float16 v4h;
typedef __attribute__((ext_vector_type(8)))  float    v8f;
typedef __attribute__((ext_vector_type(4)))  float    v4f;

// Problem dims (match reference)
constexpr int CIN = 2, COUT = 2, C = 8, F = 512, W = 384, NH = 8, L = 15, E = 4;
constexpr int D   = F / NH;          // 64
constexpr int FW  = F * W;           // 196608
constexpr int CFW = C * FW;          // 1572864
constexpr int QKN = C * NH * W * W;  // 9437184
constexpr float EPS = 1e-8f;

// Map a tile element (r16 = row/col within 16-tile, kk = 0..31) to its
// fragment slot: lane = r16 + 16*((kk>>3)&1), elem = (kk&7) + 8*(kk>>4).
// Documented CDNA5 16-bit A/B operand layout: each lane's 16 halves are
// CONTIGUOUS in LDS and load as one 32-byte v16h (2x ds_load_b128).
// For fixed r16, 4 k's within one octet (k%4==0) map to 4 CONTIGUOUS halves.
__device__ __forceinline__ int frag_idx(int r16, int kk) {
  return ((r16 + (((kk >> 3) & 1) << 4)) << 4) + (kk & 7) + ((kk >> 4) << 3);
}

// ---------------------------------------------------------------------------
// Generic batched WMMA GEMM:  Out[bi] = alpha * A[bi] x B[bi] (+ Add[bi])
//   A is MxK   (AT: addr = k*lda + m; else m*lda + k)
//   B is KxN   (BT: addr = n*ldb + k; else k*ldb + n)
//   Out MxN    (OT: addr = n*ldc + m; else m*ldc + n)   Add always row-major.
// block = 256 threads = 8 waves arranged as MT (m-tiles) x NTW (n-waves),
// each wave owns exactly TPW n-tiles -> NO guards around WMMA (EXEC all 1s).
// N fixed at compile time: NN = 16*(8/MT)*TPW.  M % (16*MT)==0, K % 32==0.
// grid = (M/(16*MT), batch).
// ---------------------------------------------------------------------------
template <int MT, int TPW, bool AT, bool BT, bool OT, bool ADD>
__global__ void wmma_gemm_kernel(const float* __restrict__ A,
                                 const float* __restrict__ Bm,
                                 float* __restrict__ Out,
                                 const float* __restrict__ Add,
                                 int K, int lda, int ldb, int ldc, int ldadd,
                                 int sA, int sB, int sO, int sAdd, float alpha) {
  constexpr int NTW = 8 / MT;          // waves along N
  constexpr int M16 = 16 * MT;         // rows per block
  constexpr int NN  = 16 * NTW * TPW;  // compile-time N
  __shared__ __align__(32) _Float16 la[MT * 512];        // m-tiles, frag order
  __shared__ __align__(32) _Float16 lb[NTW * TPW * 512]; // n-tiles, frag order

  const int bi = blockIdx.y;
  A   += (size_t)bi * sA;
  Bm  += (size_t)bi * sB;
  Out += (size_t)bi * sO;
  if (ADD) Add += (size_t)bi * sAdd;

  const int m0   = blockIdx.x * M16;
  const int tid  = threadIdx.x;
  const int lane = tid & 31;
  const int wv   = tid >> 5;
  const int wvN  = wv % NTW;           // this wave's n-tile group
  const int wvM  = wv / NTW;           // this wave's m-tile

  v8f acc[TPW] = {};

  for (int k0 = 0; k0 < K; k0 += 32) {
    __syncthreads();
    // ---- stage A tile (M16 x 32), float4 global loads, f32 -> f16 ----
    for (int i4 = tid; i4 < M16 * 8; i4 += 256) {
      if (AT) {                         // contiguous along m
        int kk = i4 / (M16 / 4);
        int r4 = (i4 % (M16 / 4)) * 4;
        const float* ap = A + (size_t)(k0 + kk) * lda + (m0 + r4);
        v4f v = *reinterpret_cast<const v4f*>(ap);
        if (k0 + 32 < K) __builtin_prefetch(ap + (size_t)32 * lda, 0, 0);
#pragma unroll
        for (int u = 0; u < 4; ++u) {
          int r = r4 + u;
          la[(r >> 4) * 512 + frag_idx(r & 15, kk)] = (_Float16)v[u];
        }
      } else {                          // contiguous along k
        int r  = i4 >> 3;
        int kg = (i4 & 7) * 4;
        const float* ap = A + (size_t)(m0 + r) * lda + (k0 + kg);
        v4f v = *reinterpret_cast<const v4f*>(ap);
        if (k0 + 32 < K) __builtin_prefetch(ap + 32, 0, 0);
        v4h hx;
#pragma unroll
        for (int u = 0; u < 4; ++u) hx[u] = (_Float16)v[u];
        *reinterpret_cast<v4h*>(&la[(r >> 4) * 512 + frag_idx(r & 15, kg)]) = hx;
      }
    }
    // ---- stage B strip (32 x NN), float4 global loads ----
    if (BT) {                           // contiguous along k -> packed b64 store
#pragma unroll
      for (int i4 = tid; i4 < NN * 8; i4 += 256) {
        int n  = i4 >> 3;
        int kg = (i4 & 7) * 4;
        v4f v = *reinterpret_cast<const v4f*>(Bm + (size_t)n * ldb + (k0 + kg));
        v4h hx;
#pragma unroll
        for (int u = 0; u < 4; ++u) hx[u] = (_Float16)v[u];
        *reinterpret_cast<v4h*>(&lb[(n >> 4) * 512 + frag_idx(n & 15, kg)]) = hx;
      }
    } else {                            // contiguous along n
      constexpr int N4 = NN / 4;
#pragma unroll
      for (int i4 = tid; i4 < 32 * N4; i4 += 256) {
        int kk = i4 / N4;
        int n4 = (i4 % N4) * 4;
        v4f v = *reinterpret_cast<const v4f*>(Bm + (size_t)(k0 + kk) * ldb + n4);
#pragma unroll
        for (int u = 0; u < 4; ++u) {
          int n = n4 + u;
          lb[(n >> 4) * 512 + frag_idx(n & 15, kk)] = (_Float16)v[u];
        }
      }
    }
    __syncthreads();

    // ---- fragments: one contiguous 32B LDS read per operand per lane ----
    const v16h af = *reinterpret_cast<const v16h*>(&la[wvM * 512 + lane * 16]);
#pragma unroll
    for (int j = 0; j < TPW; ++j) {
      const int t = wvN + j * NTW;
      const v16h bf = *reinterpret_cast<const v16h*>(&lb[t * 512 + lane * 16]);
      acc[j] = __builtin_amdgcn_wmma_f32_16x16x32_f16(
          false, af, false, bf, (short)0, acc[j], false, false);
    }
  }

  // ---- epilogue: D layout = lane(l&15)=N col, VGPR r -> M = r + 8*(l>>4) ----
#pragma unroll
  for (int j = 0; j < TPW; ++j) {
    const int t   = wvN + j * NTW;
    const int col = t * 16 + (lane & 15);
#pragma unroll
    for (int r = 0; r < 8; ++r) {
      int row  = m0 + wvM * 16 + ((lane >> 4) << 3) + r;
      float vv = alpha * acc[j][r];
      if (ADD) vv += Add[(size_t)row * ldadd + col];
      size_t o = OT ? ((size_t)col * ldc + row) : ((size_t)row * ldc + col);
      Out[o] = vv;
    }
  }
}

// ---------------------------------------------------------------------------
// 3x3 same-padded conv over [Cin,F,W] -> [gridDim.y, F, W]; optional squared
// ReLU on the output and optional fused residual add.
// ---------------------------------------------------------------------------
__global__ void conv3x3_kernel(const float* __restrict__ in, const float* __restrict__ wt,
                               const float* __restrict__ res, float* __restrict__ out,
                               int Cin, int sqrelu) {
  int p = blockIdx.x * blockDim.x + threadIdx.x;
  if (p >= FW) return;
  int oc = blockIdx.y;
  int f = p / W, w = p % W;
  float acc = 0.f;
  for (int ic = 0; ic < Cin; ++ic) {
    const float* ip = in + (size_t)ic * FW;
    const float* wp = wt + (size_t)(oc * Cin + ic) * 9;
#pragma unroll
    for (int ky = 0; ky < 3; ++ky) {
      int ff = f + ky - 1;
      if (ff < 0 || ff >= F) continue;
#pragma unroll
      for (int kx = 0; kx < 3; ++kx) {
        int ww = w + kx - 1;
        if (ww < 0 || ww >= W) continue;
        acc += wp[ky * 3 + kx] * ip[(size_t)ff * W + ww];
      }
    }
  }
  if (sqrelu) { acc = fmaxf(acc, 0.f); acc *= acc; }
  if (res) acc += res[(size_t)oc * FW + p];
  out[(size_t)oc * FW + p] = acc;
}

// LayerNorm over the F (bin) axis per (c, w); one thread per (c,w) row.
__global__ void ln_kernel(const float* __restrict__ x, const float* __restrict__ wn,
                          const float* __restrict__ bn, float* __restrict__ out) {
  int idx = blockIdx.x * blockDim.x + threadIdx.x;
  if (idx >= C * W) return;
  int c = idx / W, w = idx % W;
  const float* xp = x + (size_t)c * FW + w;
  float s = 0.f, s2 = 0.f;
  for (int f = 0; f < F; ++f) { float v = xp[(size_t)f * W]; s += v; s2 += v * v; }
  float mu  = s * (1.f / F);
  float var = s2 * (1.f / F) - mu * mu;
  float inv = rsqrtf(var + EPS);
  float* op = out + (size_t)c * FW + w;
  for (int f = 0; f < F; ++f) {
    float v = (xp[(size_t)f * W] - mu) * inv;
    op[(size_t)f * W] = v * wn[c * F + f] + bn[c * F + f];
  }
}

// Depthwise channel mix: out[d,p] = sum_c dw[d,c] * x[c,p]
__global__ void chmix_kernel(const float* __restrict__ x, const float* __restrict__ dw,
                             float* __restrict__ out) {
  int p = blockIdx.x * blockDim.x + threadIdx.x;
  if (p >= FW) return;
  float xin[C];
#pragma unroll
  for (int c = 0; c < C; ++c) xin[c] = x[(size_t)c * FW + p];
#pragma unroll
  for (int d = 0; d < C; ++d) {
    float a = 0.f;
#pragma unroll
    for (int c = 0; c < C; ++c) a += dw[d * C + c] * xin[c];
    out[(size_t)d * FW + p] = a;
  }
}

// (1,3) conv along W with channel mixing: weights [C,C,1,3], pad 1 in W.
__global__ void conv13_kernel(const float* __restrict__ x, const float* __restrict__ cw,
                              float* __restrict__ out) {
  int p = blockIdx.x * blockDim.x + threadIdx.x;
  if (p >= FW) return;
  int w = p % W;
  float xin[C][3];
#pragma unroll
  for (int c = 0; c < C; ++c)
#pragma unroll
    for (int t = 0; t < 3; ++t) {
      int ww = w + t - 1;
      xin[c][t] = (ww >= 0 && ww < W) ? x[(size_t)c * FW + p + (t - 1)] : 0.f;
    }
#pragma unroll
  for (int d = 0; d < C; ++d) {
    float a = 0.f;
#pragma unroll
    for (int c = 0; c < C; ++c)
#pragma unroll
      for (int t = 0; t < 3; ++t) a += cw[(d * C + c) * 3 + t] * xin[c][t];
    out[(size_t)d * FW + p] = a;
  }
}

// Row softmax over length W=384; one wave32 per row (12 elems/lane).
__global__ void softmax_kernel(const float* __restrict__ qk, float* __restrict__ P) {
  int row  = blockIdx.x * (blockDim.x >> 5) + (threadIdx.x >> 5);
  int lane = threadIdx.x & 31;
  if (row >= C * NH * W) return;
  const float* r = qk + (size_t)row * W;
  float v[12];
  float mx = -3.4e38f;
#pragma unroll
  for (int i = 0; i < 12; ++i) { v[i] = r[lane + i * 32]; mx = fmaxf(mx, v[i]); }
#pragma unroll
  for (int off = 16; off; off >>= 1) mx = fmaxf(mx, __shfl_xor(mx, off, 32));
  float s = 0.f;
#pragma unroll
  for (int i = 0; i < 12; ++i) { v[i] = __expf(v[i] - mx); s += v[i]; }
#pragma unroll
  for (int off = 16; off; off >>= 1) s += __shfl_xor(s, off, 32);
  float inv = 1.f / s;
  float* o = P + (size_t)row * W;
#pragma unroll
  for (int i = 0; i < 12; ++i) o[lane + i * 32] = v[i] * inv;
}

// Final 1x1 conv C -> COUT.
__global__ void outconv_kernel(const float* __restrict__ h, const float* __restrict__ wt,
                               float* __restrict__ out) {
  int p = blockIdx.x * blockDim.x + threadIdx.x;
  if (p >= FW) return;
#pragma unroll
  for (int o = 0; o < COUT; ++o) {
    float a = 0.f;
#pragma unroll
    for (int c = 0; c < C; ++c) a += wt[o * C + c] * h[(size_t)c * FW + p];
    out[(size_t)o * FW + p] = a;
  }
}

extern "C" void kernel_launch(void* const* d_in, const int* in_sizes, int n_in,
                              void* d_out, int out_size, void* d_ws, size_t ws_size,
                              hipStream_t stream) {
  (void)in_sizes; (void)n_in; (void)out_size; (void)ws_size;
  const float* x    = (const float*)d_in[0];
  const float* embw = (const float*)d_in[1];
  const float* outw = (const float*)d_in[2];
  const float* n1w  = (const float*)d_in[3];
  const float* n1b  = (const float*)d_in[4];
  const float* qpw  = (const float*)d_in[5];
  const float* qdw  = (const float*)d_in[6];
  const float* qcw  = (const float*)d_in[7];
  const float* kpw  = (const float*)d_in[8];
  const float* kdw  = (const float*)d_in[9];
  const float* kcw  = (const float*)d_in[10];
  const float* vpw  = (const float*)d_in[11];
  const float* vdw  = (const float*)d_in[12];
  const float* vcw  = (const float*)d_in[13];
  const float* opw  = (const float*)d_in[14];
  const float* odw  = (const float*)d_in[15];
  const float* n2w  = (const float*)d_in[16];
  const float* n2b  = (const float*)d_in[17];
  const float* f1w  = (const float*)d_in[18];
  const float* f2w  = (const float*)d_in[19];
  float* out = (float*)d_out;

  float* ws  = (float*)d_ws;
  float* h   = ws;  ws += CFW;
  float* hln = ws;  ws += CFW;
  float* t0  = ws;  ws += CFW;
  float* t1  = ws;  ws += CFW;
  float* qb  = ws;  ws += CFW;
  float* kb  = ws;  ws += CFW;
  float* vb  = ws;  ws += CFW;
  float* ab  = ws;  ws += CFW;
  float* mid = ws;  ws += (size_t)C * E * FW;
  float* qk0 = ws;  ws += QKN;
  float* qk1 = ws;  ws += QKN;
  float* Pm  = ws;  ws += QKN;

  const dim3 blk(256);
  const dim3 gP((FW + 255) / 256);
  const float inv_sqrtF = 1.0f / sqrtf((float)F);

  // Embedding conv: CIN -> C, 3x3 same padding.
  conv3x3_kernel<<<dim3(gP.x, C), blk, 0, stream>>>(x, embw, nullptr, h, CIN, 0);

  for (int l = 0; l < L; ++l) {
    const float* qk_prev = (l & 1) ? qk0 : qk1;
    float*       qk_cur  = (l & 1) ? qk1 : qk0;

    ln_kernel<<<dim3((C * W + 255) / 256), blk, 0, stream>>>(
        h, n1w + (size_t)l * C * F, n1b + (size_t)l * C * F, hln);

    // q / k / v: channel-mix -> per-channel FxF GEMM -> (1,3) conv
    const float* pws[3] = { qpw + (size_t)l * C * F * F, kpw + (size_t)l * C * F * F,
                            vpw + (size_t)l * C * F * F };
    const float* dws[3] = { qdw + (size_t)l * C * C, kdw + (size_t)l * C * C,
                            vdw + (size_t)l * C * C };
    const float* cws[3] = { qcw + (size_t)l * C * C * 3, kcw + (size_t)l * C * C * 3,
                            vcw + (size_t)l * C * C * 3 };
    float* bufs[3] = { qb, kb, vb };
    for (int j = 0; j < 3; ++j) {
      chmix_kernel<<<gP, blk, 0, stream>>>(hln, dws[j], t0);
      // per-channel: OUT[g,w] = PW[c][g,f] * X[c][f,w]; M=F, N=W=384, K=F
      wmma_gemm_kernel<1, 3, false, false, false, false>
          <<<dim3(F / 16, C), blk, 0, stream>>>(
          pws[j], t0, t1, nullptr, F, F, W, W, 0, F * F, FW, FW, 0, 1.0f);
      conv13_kernel<<<gP, blk, 0, stream>>>(t1, cws[j], bufs[j]);
    }

    // QK^T per (c,h): M=W, N=W=384, K=D; A = q-head (col-major), bias = prev qk.
    if (l == 0)
      wmma_gemm_kernel<1, 3, true, false, false, false>
          <<<dim3(W / 16, C * NH), blk, 0, stream>>>(
          qb, kb, qk_cur, nullptr, D, W, W, W, 0, D * W, D * W, W * W, 0, inv_sqrtF);
    else
      wmma_gemm_kernel<1, 3, true, false, false, true>
          <<<dim3(W / 16, C * NH), blk, 0, stream>>>(
          qb, kb, qk_cur, qk_prev, D, W, W, W, W, D * W, D * W, W * W, W * W, inv_sqrtF);

    softmax_kernel<<<dim3(C * NH * W / 8), blk, 0, stream>>>(qk_cur, Pm);

    // A = P @ V per (c,h): M=W, N=D=64, K=W; B col-major, transposed store.
    // MT=2 (32 rows/block) x 4 n-waves x 1 tile: all 8 waves busy, no guards.
    wmma_gemm_kernel<2, 1, false, true, true, false>
        <<<dim3(W / 32, C * NH), blk, 0, stream>>>(
        Pm, vb, ab, nullptr, W, W, W, W, 0, W * W, D * W, D * W, 0, 1.0f);

    // Output projection (channel-mix then FxF GEMM) with fused residual into h.
    chmix_kernel<<<gP, blk, 0, stream>>>(ab, odw + (size_t)l * C * C, t0);
    wmma_gemm_kernel<1, 3, false, false, false, true>
        <<<dim3(F / 16, C), blk, 0, stream>>>(
        opw + (size_t)l * C * F * F, t0, h, h, F, F, W, W, W, F * F, FW, FW, FW, 1.0f);

    // FFN: LN -> conv3x3(C->CE) + squared ReLU -> conv3x3(CE->C) + residual.
    ln_kernel<<<dim3((C * W + 255) / 256), blk, 0, stream>>>(
        h, n2w + (size_t)l * C * F, n2b + (size_t)l * C * F, hln);
    conv3x3_kernel<<<dim3(gP.x, C * E), blk, 0, stream>>>(
        hln, f1w + (size_t)l * C * E * C * 9, nullptr, mid, C, 1);
    conv3x3_kernel<<<dim3(gP.x, C), blk, 0, stream>>>(
        mid, f2w + (size_t)l * C * C * E * 9, h, h, C * E, 0);
  }

  outconv_kernel<<<gP, blk, 0, stream>>>(h, outw, out);
}
